// SimpleGAT_1_90374701842546
// MI455X (gfx1250) — matrix operand order repaired and checked by reading, not compiled, not run on hardware
//
#include <hip/hip_runtime.h>
#include <hip/hip_bf16.h>

typedef __attribute__((ext_vector_type(16))) _Float16 v16h;
typedef __attribute__((ext_vector_type(8)))  _Float16 v8h;
typedef __attribute__((ext_vector_type(8)))  float    v8f;

#define N_NODES 100000
#define N_EDGES 1600000
#define F_IN    128
#define EDIM    16
#define HEADS   4
#define DIM     16
#define HC      64
#define ETOT    (N_EDGES + N_NODES)

// ---------------------------------------------------------------- utilities

__global__ void fill_f32_kernel(float* __restrict__ p, float v, int n) {
  int i = blockIdx.x * blockDim.x + threadIdx.x;
  if (i < n) p[i] = v;
}

__global__ void cvt_f16_kernel(const float* __restrict__ in, _Float16* __restrict__ out, int n) {
  int i = blockIdx.x * blockDim.x + threadIdx.x;
  if (i < n) out[i] = (_Float16)in[i];
}

// ----------------------------------------------- self-loop mean edge attrs

__global__ void edge_mean_accum_kernel(const int* __restrict__ ei,
                                       const float* __restrict__ ea,
                                       float* __restrict__ loop_sum,
                                       float* __restrict__ cnt) {
  int idx = blockIdx.x * blockDim.x + threadIdx.x;
  if (idx >= N_EDGES * EDIM) return;
  int e = idx >> 4, k = idx & 15;
  int dst = ei[N_EDGES + e];
  atomicAdd(&loop_sum[dst * EDIM + k], ea[idx]);
  if (k == 0) atomicAdd(&cnt[dst], 1.0f);
}

__global__ void loop_finalize_kernel(float* __restrict__ loop_sum,
                                     const float* __restrict__ cnt) {
  int idx = blockIdx.x * blockDim.x + threadIdx.x;
  if (idx >= N_NODES * EDIM) return;
  int n = idx >> 4;
  loop_sum[idx] = loop_sum[idx] / fmaxf(cnt[n], 1.0f);
}

// ------------------------------------------------- WMMA B-fragment packing
// B 32x16 f16 fragment (ISA 7.12.2): lane(0..15)=column; element h<8 -> K 8*(lane/16)+h,
// h>=8 -> K 16+8*(lane/16)+(h-8).  out index = ((kt*4 + ntile)*32 + lane)*16 + h.

__global__ void pack_b_kernel(const float* __restrict__ W, _Float16* __restrict__ out, int K) {
  int idx = blockIdx.x * blockDim.x + threadIdx.x;
  if (idx >= K * HC) return;
  int h    = idx & 15;
  int lane = (idx >> 4) & 31;
  int nt   = (idx >> 9) & 3;
  int kt   = idx >> 11;
  int k = kt * 32 + (h >> 3) * 16 + (lane >> 4) * 8 + (h & 7);
  int n = nt * 16 + (lane & 15);
  out[idx] = (_Float16)W[k * HC + n];
}

// ------------------------------------------------------- dual WMMA GEMM
// One wave computes a 16-row tile of BOTH  Cl = A@Bl + bl  and  Cr = A@Br + br
// (A fragment reused).  A is row-major [M,K] f16, B pre-packed fragments.

template <int K>
__global__ void gemm_dual_kernel(const _Float16* __restrict__ A,
                                 const _Float16* __restrict__ Bl,
                                 const _Float16* __restrict__ Br,
                                 const float* __restrict__ bl,
                                 const float* __restrict__ br,
                                 float* __restrict__ Cl,
                                 float* __restrict__ Cr,
                                 int M) {
  const int lane  = threadIdx.x & 31;
  const int wave  = threadIdx.x >> 5;
  const int mtile = blockIdx.x * 4 + wave;
  if (mtile * 16 >= M) return;              // wave-uniform: EXEC stays all-ones
  const int m0 = mtile * 16;
  const int r  = lane & 15;
  const int hi = lane >> 4;

  v8f accl[4] = {}, accr[4] = {};

#pragma unroll
  for (int kt = 0; kt < K / 32; ++kt) {
    // A fragment: lane<16 -> K {0..7, 16..23}; lane>=16 -> K {8..15, 24..31}
    const _Float16* ap = A + (size_t)(m0 + r) * K + kt * 32 + hi * 8;
    v8h alo = *(const v8h*)(ap);
    v8h ahi = *(const v8h*)(ap + 16);
    v16h a;
#pragma unroll
    for (int i = 0; i < 8; ++i) { a[i] = alo[i]; a[8 + i] = ahi[i]; }

#pragma unroll
    for (int t = 0; t < 4; ++t) {
      v16h b0 = *(const v16h*)(Bl + ((size_t)(kt * 4 + t) * 32 + lane) * 16);
      accl[t] = __builtin_amdgcn_wmma_f32_16x16x32_f16(false, a, false, b0,
                                                       (short)0, accl[t], false, false);
      v16h b1 = *(const v16h*)(Br + ((size_t)(kt * 4 + t) * 32 + lane) * 16);
      accr[t] = __builtin_amdgcn_wmma_f32_16x16x32_f16(false, a, false, b1,
                                                       (short)0, accr[t], false, false);
    }
  }

  // C layout: lane<16 -> col=lane, rows m0+v; lane>=16 -> col=lane-16, rows m0+8+v
#pragma unroll
  for (int t = 0; t < 4; ++t) {
#pragma unroll
    for (int v = 0; v < 8; ++v) {
      int row = m0 + hi * 8 + v;
      int col = t * 16 + r;
      if (row < M) {
        Cl[(size_t)row * HC + col] = accl[t][v] + bl[col];
        Cr[(size_t)row * HC + col] = accr[t][v] + br[col];
      }
    }
  }
}

// ----------------------------------------------------------- attention passes

__device__ __forceinline__ void atomicMaxFloat(float* addr, float v) {
  if (v >= 0.0f) atomicMax((int*)addr, __float_as_int(v));
  else           atomicMin((unsigned int*)addr, __float_as_uint(v));
}

// pass 1: alpha[e,h] = sum_c att[h,c] * leaky_relu(xl[src]+xr[dst]+ (ea@We)[h,c])
__global__ void attn_alpha_kernel(const float* __restrict__ xl, const float* __restrict__ xr,
                                  const int* __restrict__ ei, const float* __restrict__ ea,
                                  const float* __restrict__ loop_attr,
                                  const float* __restrict__ We, const float* __restrict__ att,
                                  float* __restrict__ alpha, float* __restrict__ amax) {
  int idx = blockIdx.x * blockDim.x + threadIdx.x;
  if (idx >= ETOT * HEADS) return;
  int e = idx >> 2, h = idx & 3;
  int src, dst;
  const float* arow;
  if (e < N_EDGES) {
    src = ei[e]; dst = ei[N_EDGES + e];
    arow = ea + (size_t)e * EDIM;
  } else {
    src = dst = e - N_EDGES;
    arow = loop_attr + (size_t)src * EDIM;
  }
  float av[EDIM];
#pragma unroll
  for (int k = 0; k < EDIM; ++k) av[k] = arow[k];

  const float* xls  = xl + (size_t)src * HC + h * DIM;
  const float* xrd  = xr + (size_t)dst * HC + h * DIM;
  const float* atth = att + h * DIM;
  float acc = 0.0f;
#pragma unroll
  for (int c = 0; c < DIM; ++c) {
    float ee = 0.0f;
#pragma unroll
    for (int k = 0; k < EDIM; ++k) ee += av[k] * We[k * HC + h * DIM + c];
    float s = xls[c] + xrd[c] + ee;
    s = (s > 0.0f) ? s : 0.2f * s;     // leaky_relu(0.2)
    acc += s * atth[c];
  }
  alpha[idx] = acc;
  atomicMaxFloat(&amax[dst * HEADS + h], acc);
}

// pass 2: ex = exp(alpha - amax[dst]); denom[dst] += ex  (alpha overwritten)
__global__ void attn_exp_kernel(float* __restrict__ alpha, const float* __restrict__ amax,
                                float* __restrict__ denom, const int* __restrict__ ei) {
  int idx = blockIdx.x * blockDim.x + threadIdx.x;
  if (idx >= ETOT * HEADS) return;
  int e = idx >> 2, h = idx & 3;
  int dst = (e < N_EDGES) ? ei[N_EDGES + e] : (e - N_EDGES);
  float ex = expf(alpha[idx] - amax[dst * HEADS + h]);
  alpha[idx] = ex;
  atomicAdd(&denom[dst * HEADS + h], ex);
}

// pass 3: acc[dst] += xl[src] * ex / (denom[dst] + 1e-16)
__global__ void attn_scatter_kernel(const float* __restrict__ alpha,
                                    const float* __restrict__ denom,
                                    const float* __restrict__ xl,
                                    const int* __restrict__ ei,
                                    float* __restrict__ acc) {
  int idx = blockIdx.x * blockDim.x + threadIdx.x;
  if (idx >= ETOT * HEADS) return;
  int e = idx >> 2, h = idx & 3;
  int src, dst;
  if (e < N_EDGES) { src = ei[e]; dst = ei[N_EDGES + e]; }
  else             { src = dst = e - N_EDGES; }
  float a = alpha[idx] / (denom[dst * HEADS + h] + 1e-16f);
  const float* xls = xl + (size_t)src * HC + h * DIM;
  float* op = acc + (size_t)dst * HC + h * DIM;
#pragma unroll
  for (int c = 0; c < DIM; ++c) atomicAdd(&op[c], xls[c] * a);
}

// h = relu(acc + bias) -> f16 (input for next layer's WMMA GEMM)
__global__ void bias_relu_h16_kernel(const float* __restrict__ acc,
                                     const float* __restrict__ bias,
                                     _Float16* __restrict__ h16) {
  int idx = blockIdx.x * blockDim.x + threadIdx.x;
  if (idx >= N_NODES * HC) return;
  float v = acc[idx] + bias[idx & 63];
  h16[idx] = (_Float16)(v > 0.0f ? v : 0.0f);
}

// final: relu(h2@W_lin1+b) -> @W_lin2+b -> sigmoid
__global__ void final_mlp_kernel(const float* __restrict__ acc, const float* __restrict__ bias2,
                                 const float* __restrict__ Wl, const float* __restrict__ blin,
                                 const float* __restrict__ Wo, const float* __restrict__ bo,
                                 float* __restrict__ out) {
  int n = blockIdx.x * blockDim.x + threadIdx.x;
  if (n >= N_NODES) return;
  float hrow[HC];
#pragma unroll
  for (int j = 0; j < HC; ++j) {
    float v = acc[(size_t)n * HC + j] + bias2[j];
    hrow[j] = v > 0.0f ? v : 0.0f;
  }
  float z = bo[0];
#pragma unroll
  for (int c = 0; c < DIM; ++c) {
    float t = blin[c];
    for (int j = 0; j < HC; ++j) t += hrow[j] * Wl[j * DIM + c];
    t = t > 0.0f ? t : 0.0f;
    z += t * Wo[c];
  }
  out[n] = 1.0f / (1.0f + expf(-z));
}

// ---------------------------------------------------------------- launcher

extern "C" void kernel_launch(void* const* d_in, const int* in_sizes, int n_in,
                              void* d_out, int out_size, void* d_ws, size_t ws_size,
                              hipStream_t stream) {
  const float* x     = (const float*)d_in[0];
  const int*   ei    = (const int*)d_in[1];
  const float* ea    = (const float*)d_in[2];
  const float* Wl1   = (const float*)d_in[3];
  const float* bl1   = (const float*)d_in[4];
  const float* Wr1   = (const float*)d_in[5];
  const float* br1   = (const float*)d_in[6];
  const float* We1   = (const float*)d_in[7];
  const float* att1  = (const float*)d_in[8];
  const float* bias1 = (const float*)d_in[9];
  const float* Wl2   = (const float*)d_in[10];
  const float* bl2   = (const float*)d_in[11];
  const float* Wr2   = (const float*)d_in[12];
  const float* br2   = (const float*)d_in[13];
  const float* We2   = (const float*)d_in[14];
  const float* att2  = (const float*)d_in[15];
  const float* bias2 = (const float*)d_in[16];
  const float* W_l1  = (const float*)d_in[17];
  const float* b_l1  = (const float*)d_in[18];
  const float* W_l2  = (const float*)d_in[19];
  const float* b_l2  = (const float*)d_in[20];

  char* ws = (char*)d_ws;
  size_t off = 0;
  auto take = [&](size_t bytes) -> char* {
    char* p = ws + off;
    off = (off + bytes + 255) & ~(size_t)255;
    return p;
  };

  _Float16* xh    = (_Float16*)take((size_t)N_NODES * F_IN * 2);
  _Float16* Wl1p  = (_Float16*)take((size_t)F_IN * HC * 2);
  _Float16* Wr1p  = (_Float16*)take((size_t)F_IN * HC * 2);
  _Float16* Wl2p  = (_Float16*)take((size_t)HC * HC * 2);
  _Float16* Wr2p  = (_Float16*)take((size_t)HC * HC * 2);
  float*    xl    = (float*)take((size_t)N_NODES * HC * 4);
  float*    xr    = (float*)take((size_t)N_NODES * HC * 4);
  float*    loopb = (float*)take((size_t)N_NODES * EDIM * 4);
  float*    cnt   = (float*)take((size_t)N_NODES * 4);
  float*    alpha = (float*)take((size_t)ETOT * HEADS * 4);
  float*    amax  = (float*)take((size_t)N_NODES * HEADS * 4);
  float*    denom = (float*)take((size_t)N_NODES * HEADS * 4);
  float*    accb  = (float*)take((size_t)N_NODES * HC * 4);
  _Float16* h16   = (_Float16*)take((size_t)N_NODES * HC * 2);

  const int TB = 256;
  auto nblk = [&](long n) { return dim3((unsigned)((n + TB - 1) / TB)); };
  const dim3 gemm_grid((N_NODES / 16 + 3) / 4);   // 6250 M-tiles, 4 waves/block

  // ---- self-loop mean edge attrs
  fill_f32_kernel<<<nblk((long)N_NODES * EDIM), TB, 0, stream>>>(loopb, 0.0f, N_NODES * EDIM);
  fill_f32_kernel<<<nblk(N_NODES), TB, 0, stream>>>(cnt, 0.0f, N_NODES);
  edge_mean_accum_kernel<<<nblk((long)N_EDGES * EDIM), TB, 0, stream>>>(ei, ea, loopb, cnt);
  loop_finalize_kernel<<<nblk((long)N_NODES * EDIM), TB, 0, stream>>>(loopb, cnt);

  // ---- precision conversion + B-fragment packing
  cvt_f16_kernel<<<nblk((long)N_NODES * F_IN), TB, 0, stream>>>(x, xh, N_NODES * F_IN);
  pack_b_kernel<<<nblk(F_IN * HC), TB, 0, stream>>>(Wl1, Wl1p, F_IN);
  pack_b_kernel<<<nblk(F_IN * HC), TB, 0, stream>>>(Wr1, Wr1p, F_IN);
  pack_b_kernel<<<nblk(HC * HC), TB, 0, stream>>>(Wl2, Wl2p, HC);
  pack_b_kernel<<<nblk(HC * HC), TB, 0, stream>>>(Wr2, Wr2p, HC);

  // ================= layer 1 =================
  gemm_dual_kernel<F_IN><<<gemm_grid, 128, 0, stream>>>(xh, Wl1p, Wr1p, bl1, br1, xl, xr, N_NODES);

  fill_f32_kernel<<<nblk(N_NODES * HEADS), TB, 0, stream>>>(amax, -3.402823e38f, N_NODES * HEADS);
  fill_f32_kernel<<<nblk(N_NODES * HEADS), TB, 0, stream>>>(denom, 0.0f, N_NODES * HEADS);
  fill_f32_kernel<<<nblk((long)N_NODES * HC), TB, 0, stream>>>(accb, 0.0f, N_NODES * HC);

  attn_alpha_kernel<<<nblk((long)ETOT * HEADS), TB, 0, stream>>>(xl, xr, ei, ea, loopb, We1, att1, alpha, amax);
  attn_exp_kernel<<<nblk((long)ETOT * HEADS), TB, 0, stream>>>(alpha, amax, denom, ei);
  attn_scatter_kernel<<<nblk((long)ETOT * HEADS), TB, 0, stream>>>(alpha, denom, xl, ei, accb);
  bias_relu_h16_kernel<<<nblk((long)N_NODES * HC), TB, 0, stream>>>(accb, bias1, h16);

  // ================= layer 2 =================
  gemm_dual_kernel<HC><<<gemm_grid, 128, 0, stream>>>(h16, Wl2p, Wr2p, bl2, br2, xl, xr, N_NODES);

  fill_f32_kernel<<<nblk(N_NODES * HEADS), TB, 0, stream>>>(amax, -3.402823e38f, N_NODES * HEADS);
  fill_f32_kernel<<<nblk(N_NODES * HEADS), TB, 0, stream>>>(denom, 0.0f, N_NODES * HEADS);
  fill_f32_kernel<<<nblk((long)N_NODES * HC), TB, 0, stream>>>(accb, 0.0f, N_NODES * HC);

  attn_alpha_kernel<<<nblk((long)ETOT * HEADS), TB, 0, stream>>>(xl, xr, ei, ea, loopb, We2, att2, alpha, amax);
  attn_exp_kernel<<<nblk((long)ETOT * HEADS), TB, 0, stream>>>(alpha, amax, denom, ei);
  attn_scatter_kernel<<<nblk((long)ETOT * HEADS), TB, 0, stream>>>(alpha, denom, xl, ei, accb);

  // ---- final MLP + sigmoid
  final_mlp_kernel<<<nblk(N_NODES), TB, 0, stream>>>(accb, bias2, W_l1, b_l1, W_l2, b_l2, (float*)d_out);
}
